// SSMLayer_25675314495527
// MI455X (gfx1250) — compile-verified, matrix operand order checked by reading, not run on hardware
//
#include <hip/hip_runtime.h>
#include <hip/hip_bf16.h>

// ---------------------------------------------------------------------------
// Mamba SSM layer for gfx1250 (MI455X), wave32 + WMMA bf16.
//  - all GEMMs consume pre-converted bf16 operands (no cvt in the K loop)
//  - B tile double-buffered in LDS (1 barrier per K step), ds_load_b128 reads
//  - per wave: 2 A fragments x NT/16 B fragments -> 2*NT/16 WMMAs per K step
//    (each LDS B fragment feeds 2 WMMAs; each A global byte feeds NT/16)
//  - chunked diagonal-state scan (3 passes) for parallelism over L
// ---------------------------------------------------------------------------

typedef __attribute__((ext_vector_type(16))) __bf16 v16bf;
typedef __attribute__((ext_vector_type(8)))  __bf16 v8bf;
typedef __attribute__((ext_vector_type(4)))  __bf16 v4bf;
typedef __attribute__((ext_vector_type(8)))  float  v8f;

// Problem constants
constexpr int BATCH   = 2;
constexpr int SEQ     = 2048;
constexpr int D_MODEL = 1024;
constexpr int D_INNER = 2048;
constexpr int D_STATE = 16;
constexpr int DT_RANK = 64;
constexpr int NX      = DT_RANK + 2 * D_STATE;   // 96
constexpr int MROWS   = BATCH * SEQ;             // 4096
constexpr int NCHUNK  = 32;                      // chunks along L
constexpr int CLEN    = SEQ / NCHUNK;            // 64

__device__ __forceinline__ __bf16 tobf(float f) { return (__bf16)f; }

__device__ __forceinline__ float softplus_f(float x) {
    return x > 20.f ? x : log1pf(__expf(x));
}

// One lane's 16-element bf16 fragment: K = kbase+j (j<8), kbase+16+(j-8) (j>=8).
// p already includes kbase. Works for both global and LDS pointers.
template <typename PT>
__device__ __forceinline__ v16bf frag_bf16(PT p) {
    v8bf lo = *(const v8bf*)(p);
    v8bf hi = *(const v8bf*)(p + 16);
    v16bf r;
#pragma unroll
    for (int i = 0; i < 8; ++i) { r[i] = lo[i]; r[8 + i] = hi[i]; }
    return r;
}

// ---------------------------------------------------------------------------
// Elementwise f32 -> bf16 (8 elements/thread, vector loads/stores)
// ---------------------------------------------------------------------------
__global__ __launch_bounds__(256) void f32_to_bf16_kernel(
    const float* __restrict__ src, __bf16* __restrict__ dst)
{
    size_t i = ((size_t)blockIdx.x * blockDim.x + threadIdx.x) * 8;
    float4 a = *(const float4*)(src + i);
    float4 b = *(const float4*)(src + i + 4);
    v8bf r;
    r[0] = tobf(a.x); r[1] = tobf(a.y); r[2] = tobf(a.z); r[3] = tobf(a.w);
    r[4] = tobf(b.x); r[5] = tobf(b.y); r[6] = tobf(b.z); r[7] = tobf(b.w);
    *(v8bf*)(dst + i) = r;
}

// ---------------------------------------------------------------------------
// WMMA bf16 GEMM:  C[M,N] = A[M,K] * W[N,K]^T
//   MODE 0: fp32 store   MODE 1: softplus(x + bias[n])   MODE 2: fp32 + bf16
//   NT: block N-tile (32 or 64)
// Block: 256 threads = 8 waves; block tile 256M x NT-N; wave tile 32M x NT-N.
// B tile (NT x 32 bf16) double-buffered in LDS: one barrier per K step.
// ---------------------------------------------------------------------------
template <int MODE, int NT>
__global__ __launch_bounds__(256) void gemm_wmma(
    const __bf16* __restrict__ Ap, const __bf16* __restrict__ Wp,
    float* __restrict__ Cp, __bf16* __restrict__ Cb,
    const float* __restrict__ bias, int K, int lda, int ldb, int ldc)
{
    constexpr int NACC = NT / 16;     // B fragments / N accumulators per row-frag
    __shared__ __bf16 Bs[2 * NT * 32];

    const int tid   = threadIdx.x;
    const int wave  = tid >> 5;
    const int lane  = tid & 31;
    const int row   = lane & 15;
    const int hsel  = lane >> 4;
    const int kbase = hsel * 8;
    const int m0    = blockIdx.y * 256 + wave * 32;   // 2 row-fragments per wave
    const int n0    = blockIdx.x * NT;

    // B staging: NT*32 elements / 256 threads = NT/8 contiguous bf16 each
    constexpr int CH  = NT / 8;       // 4 (NT=32) or 8 (NT=64)
    constexpr int TPR = 32 / CH;      // threads per tile row
    const int sn = tid / TPR;
    const int sk = (tid % TPR) * CH;
    const __bf16* wsrc  = Wp + (size_t)(n0 + sn) * ldb + sk;
    const __bf16* asrc0 = Ap + (size_t)(m0 + row) * lda + kbase;
    const __bf16* asrc1 = asrc0 + (size_t)16 * lda;

    v8f acc0[NACC]; v8f acc1[NACC];
#pragma unroll
    for (int j = 0; j < NACC; ++j) { acc0[j] = (v8f){}; acc1[j] = (v8f){}; }

    for (int kk = 0, it = 0; kk < K; kk += 32, ++it) {
        __bf16* buf = Bs + (it & 1) * (NT * 32);

        v16bf a0 = frag_bf16(asrc0 + kk);             // A direct global->VGPR
        v16bf a1 = frag_bf16(asrc1 + kk);
        if (kk + 32 < K) {                            // -> global_prefetch_b8
            __builtin_prefetch(wsrc  + kk + 32, 0, 1);
            __builtin_prefetch(asrc0 + kk + 32, 0, 1);
            __builtin_prefetch(asrc1 + kk + 32, 0, 1);
        }
        if constexpr (CH == 8) {
            v8bf wb = *(const v8bf*)(wsrc + kk);
            *(v8bf*)(buf + sn * 32 + sk) = wb;
        } else {
            v4bf wb = *(const v4bf*)(wsrc + kk);
            *(v4bf*)(buf + sn * 32 + sk) = wb;
        }
        // Single barrier per step: stores to buffer (it&1) are ordered after
        // everyone's reads of it (2 steps ago) by the intervening barrier.
        __syncthreads();

        // Batch all B-fragment LDS loads, then run the WMMA chain back-to-back.
        v16bf bfr[NACC];
#pragma unroll
        for (int j = 0; j < NACC; ++j)
            bfr[j] = frag_bf16(buf + (j * 16 + row) * 32 + kbase);
#pragma unroll
        for (int j = 0; j < NACC; ++j) {
            acc0[j] = __builtin_amdgcn_wmma_f32_16x16x32_bf16(
                false, a0, false, bfr[j], (short)0, acc0[j], false, false);
            acc1[j] = __builtin_amdgcn_wmma_f32_16x16x32_bf16(
                false, a1, false, bfr[j], (short)0, acc1[j], false, false);
        }
    }

#pragma unroll
    for (int j = 0; j < NACC; ++j) {
#pragma unroll
        for (int i = 0; i < 8; ++i) {
            int r0  = m0 + hsel * 8 + i;              // D layout: VGPR i <-> M row
            int r1  = r0 + 16;
            int col = n0 + j * 16 + row;
            float v0 = acc0[j][i];
            float v1 = acc1[j][i];
            if (MODE == 1) {
                v0 = softplus_f(v0 + bias[col]);
                v1 = softplus_f(v1 + bias[col]);
            }
            Cp[(size_t)r0 * ldc + col] = v0;
            Cp[(size_t)r1 * ldc + col] = v1;
            if (MODE == 2) {
                Cb[(size_t)r0 * ldc + col] = tobf(v0);
                Cb[(size_t)r1 * ldc + col] = tobf(v1);
            }
        }
    }
}

// ---------------------------------------------------------------------------
// Causal depthwise conv1d (K=4) + SiLU; writes u in f32 (scan) and bf16 (GEMM)
// ---------------------------------------------------------------------------
__global__ __launch_bounds__(256) void conv_silu_kernel(
    const float* __restrict__ xz, const float* __restrict__ cw,
    const float* __restrict__ cb, float* __restrict__ u,
    __bf16* __restrict__ ubf)
{
    int gid = blockIdx.x * blockDim.x + threadIdx.x;
    int d = gid % D_INNER;
    int l = (gid / D_INNER) % SEQ;
    int b = gid / (D_INNER * SEQ);
    float acc = cb[d];
#pragma unroll
    for (int j = 0; j < 4; ++j) {
        int t = l + j - 3;
        if (t >= 0)
            acc += cw[d * 4 + j] * xz[((size_t)(b * SEQ + t)) * (2 * D_INNER) + d];
    }
    float s = acc / (1.f + __expf(-acc));
    u[gid]   = s;
    ubf[gid] = tobf(s);
}

// ---------------------------------------------------------------------------
// Selective scan, chunked (diagonal state: chunk composition is elementwise)
// ---------------------------------------------------------------------------
__global__ __launch_bounds__(256) void scan_local_kernel(
    const float* __restrict__ dt_sp, const float* __restrict__ u,
    const float* __restrict__ xdbl,  const float* __restrict__ A_log,
    float* __restrict__ Hend, float* __restrict__ P)
{
    int gid = blockIdx.x * blockDim.x + threadIdx.x;
    int d = gid % D_INNER;
    int c = (gid / D_INNER) % NCHUNK;
    int b = gid / (D_INNER * NCHUNK);

    float Ar[D_STATE];
#pragma unroll
    for (int s = 0; s < D_STATE; ++s) Ar[s] = -__expf(A_log[d * D_STATE + s]);

    float h[D_STATE]; float p[D_STATE];
#pragma unroll
    for (int s = 0; s < D_STATE; ++s) { h[s] = 0.f; p[s] = 1.f; }

    int t0 = c * CLEN;
    for (int i = 0; i < CLEN; ++i) {
        size_t bt = (size_t)(b * SEQ + t0 + i);
        float dtv = dt_sp[bt * D_INNER + d];
        float uv  = u[bt * D_INNER + d];
        float du  = dtv * uv;
        const float* bc = xdbl + bt * NX;     // cols 64..79 = B_t
#pragma unroll
        for (int s = 0; s < D_STATE; ++s) {
            float dA = __expf(dtv * Ar[s]);
            h[s] = dA * h[s] + du * bc[DT_RANK + s];
            p[s] *= dA;
        }
    }
    size_t base = (((size_t)b * NCHUNK + c) * D_STATE) * D_INNER + d;
#pragma unroll
    for (int s = 0; s < D_STATE; ++s) {
        Hend[base + (size_t)s * D_INNER] = h[s];
        P[base + (size_t)s * D_INNER]    = p[s];
    }
}

__global__ __launch_bounds__(256) void scan_prefix_kernel(
    const float* __restrict__ Hend, const float* __restrict__ P,
    float* __restrict__ H0)
{
    int gid = blockIdx.x * blockDim.x + threadIdx.x;
    int d = gid % D_INNER;
    int s = (gid / D_INNER) % D_STATE;
    int b = gid / (D_INNER * D_STATE);
    float carry = 0.f;
    for (int c = 0; c < NCHUNK; ++c) {
        size_t idx = (((size_t)b * NCHUNK + c) * D_STATE + s) * D_INNER + d;
        H0[idx] = carry;
        carry = P[idx] * carry + Hend[idx];
    }
}

__global__ __launch_bounds__(256) void scan_apply_kernel(
    const float* __restrict__ dt_sp, const float* __restrict__ u,
    const float* __restrict__ xdbl,  const float* __restrict__ A_log,
    const float* __restrict__ H0,    const float* __restrict__ Dp,
    const float* __restrict__ xz,    __bf16* __restrict__ yg)
{
    int gid = blockIdx.x * blockDim.x + threadIdx.x;
    int d = gid % D_INNER;
    int c = (gid / D_INNER) % NCHUNK;
    int b = gid / (D_INNER * NCHUNK);

    float Ar[D_STATE];
#pragma unroll
    for (int s = 0; s < D_STATE; ++s) Ar[s] = -__expf(A_log[d * D_STATE + s]);

    float h[D_STATE];
    size_t base = (((size_t)b * NCHUNK + c) * D_STATE) * D_INNER + d;
#pragma unroll
    for (int s = 0; s < D_STATE; ++s) h[s] = H0[base + (size_t)s * D_INNER];

    float Dd = Dp[d];
    int t0 = c * CLEN;
    for (int i = 0; i < CLEN; ++i) {
        size_t bt = (size_t)(b * SEQ + t0 + i);
        float dtv = dt_sp[bt * D_INNER + d];
        float uv  = u[bt * D_INNER + d];
        float du  = dtv * uv;
        const float* bc = xdbl + bt * NX;
        float y = 0.f;
#pragma unroll
        for (int s = 0; s < D_STATE; ++s) {
            float dA = __expf(dtv * Ar[s]);
            h[s] = dA * h[s] + du * bc[DT_RANK + s];       // B_t
            y   += h[s] * bc[DT_RANK + D_STATE + s];       // C_t
        }
        y += Dd * uv;
        float z = xz[bt * (2 * D_INNER) + D_INNER + d];
        float g = z / (1.f + __expf(-z));                  // silu gate
        yg[bt * D_INNER + d] = tobf(y * g);
    }
}

// ---------------------------------------------------------------------------
// Host-side launch
// ---------------------------------------------------------------------------
extern "C" void kernel_launch(void* const* d_in, const int* in_sizes, int n_in,
                              void* d_out, int out_size, void* d_ws, size_t ws_size,
                              hipStream_t stream) {
    const float* x      = (const float*)d_in[0];
    const float* W_in   = (const float*)d_in[1];
    const float* conv_w = (const float*)d_in[2];
    const float* conv_b = (const float*)d_in[3];
    const float* W_xprj = (const float*)d_in[4];
    const float* W_dt   = (const float*)d_in[5];
    const float* b_dt   = (const float*)d_in[6];
    const float* A_log  = (const float*)d_in[7];
    const float* Dp     = (const float*)d_in[8];
    const float* W_out  = (const float*)d_in[9];
    float* out = (float*)d_out;

    char* ws = (char*)d_ws;
    size_t off = 0;
    auto carve = [&](size_t bytes) { size_t o = off; off += (bytes + 255) & ~(size_t)255; return o; };

    float*  xz      = (float*) (ws + carve((size_t)MROWS * 2 * D_INNER * 4));  // 64 MB
    float*  u       = (float*) (ws + carve((size_t)MROWS * D_INNER * 4));      // 32 MB
    float*  xdbl    = (float*) (ws + carve((size_t)MROWS * NX * 4));           // 1.5 MB
    float*  dt_sp   = (float*) (ws + carve((size_t)MROWS * D_INNER * 4));      // 32 MB
    float*  Hend    = (float*) (ws + carve((size_t)BATCH * NCHUNK * D_STATE * D_INNER * 4));
    float*  Pbuf    = (float*) (ws + carve((size_t)BATCH * NCHUNK * D_STATE * D_INNER * 4));
    float*  H0      = (float*) (ws + carve((size_t)BATCH * NCHUNK * D_STATE * D_INNER * 4));
    __bf16* yg      = (__bf16*)(ws + carve((size_t)MROWS * D_INNER * 2));      // 16 MB
    __bf16* xbf     = (__bf16*)(ws + carve((size_t)MROWS * D_MODEL * 2));      // 8 MB
    __bf16* Win_bf  = (__bf16*)(ws + carve((size_t)2 * D_INNER * D_MODEL * 2));// 8 MB
    __bf16* Wxp_bf  = (__bf16*)(ws + carve((size_t)NX * D_INNER * 2));
    __bf16* Wdt_bf  = (__bf16*)(ws + carve((size_t)D_INNER * DT_RANK * 2));
    __bf16* Wout_bf = (__bf16*)(ws + carve((size_t)D_MODEL * D_INNER * 2));
    __bf16* ubf     = (__bf16*)(ws + carve((size_t)MROWS * D_INNER * 2));      // 16 MB
    __bf16* xdbl_bf = (__bf16*)(ws + carve((size_t)MROWS * NX * 2));

    // 0) one-time f32 -> bf16 conversions (activation x and all weights)
    auto cvt = [&](const float* s, __bf16* d, size_t n) {
        f32_to_bf16_kernel<<<n / (256 * 8), 256, 0, stream>>>(s, d);
    };
    cvt(x,      xbf,     (size_t)MROWS * D_MODEL);
    cvt(W_in,   Win_bf,  (size_t)2 * D_INNER * D_MODEL);
    cvt(W_xprj, Wxp_bf,  (size_t)NX * D_INNER);
    cvt(W_dt,   Wdt_bf,  (size_t)D_INNER * DT_RANK);
    cvt(W_out,  Wout_bf, (size_t)D_MODEL * D_INNER);

    // 1) in_proj: xz[4096,4096] = x @ W_in^T
    gemm_wmma<0, 64><<<dim3(2 * D_INNER / 64, MROWS / 256), 256, 0, stream>>>(
        xbf, Win_bf, xz, nullptr, nullptr, D_MODEL, D_MODEL, D_MODEL, 2 * D_INNER);

    // 2) causal depthwise conv + SiLU -> u (f32 + bf16)
    conv_silu_kernel<<<(size_t)MROWS * D_INNER / 256, 256, 0, stream>>>(
        xz, conv_w, conv_b, u, ubf);

    // 3) x_proj: xdbl[4096,96] = u @ W_xproj^T  (f32 + bf16 mirror; N=96 -> NT=32)
    gemm_wmma<2, 32><<<dim3(NX / 32, MROWS / 256), 256, 0, stream>>>(
        ubf, Wxp_bf, xdbl, xdbl_bf, nullptr, D_INNER, D_INNER, D_INNER, NX);

    // 4) dt: dt_sp[4096,2048] = softplus(xdbl[:, :64] @ W_dt^T + b_dt)
    gemm_wmma<1, 64><<<dim3(D_INNER / 64, MROWS / 256), 256, 0, stream>>>(
        xdbl_bf, Wdt_bf, dt_sp, nullptr, b_dt, DT_RANK, NX, DT_RANK, D_INNER);

    // 5) chunked selective scan
    scan_local_kernel<<<BATCH * D_INNER * NCHUNK / 256, 256, 0, stream>>>(
        dt_sp, u, xdbl, A_log, Hend, Pbuf);
    scan_prefix_kernel<<<BATCH * D_INNER * D_STATE / 256, 256, 0, stream>>>(
        Hend, Pbuf, H0);
    scan_apply_kernel<<<BATCH * D_INNER * NCHUNK / 256, 256, 0, stream>>>(
        dt_sp, u, xdbl, A_log, H0, Dp, xz, yg);

    // 6) out_proj: out[4096,1024] = yg(bf16) @ W_out^T
    gemm_wmma<0, 64><<<dim3(D_MODEL / 64, MROWS / 256), 256, 0, stream>>>(
        yg, Wout_bf, out, nullptr, nullptr, D_INNER, D_INNER, D_INNER, D_MODEL);
}
